// SurfaceDiceLoss_51866025066905
// MI455X (gfx1250) — compile-verified, matrix operand order checked by reading, not run on hardware
//
#include <hip/hip_runtime.h>
#include <hip/hip_bf16.h>

typedef __attribute__((ext_vector_type(16))) _Float16 v16h;
typedef __attribute__((ext_vector_type(8)))  float    v8f;

#define ZD 16
#define YD 384
#define XD 384
#define NPOS (17 * 385 * 385)            // 2,519,825 corner positions
#define NGROUPS32 ((NPOS + 31) / 32)     // 32 positions per wave pass
#define SMOOTH 1e-3f
#define EPSLOG 1e-12f
#define FLTBIG 3.4e38f

union V16HU { v16h h; unsigned int u[8]; };

// ---------------------------------------------------------------------------
// Fused dual 256-code argmax of bw via the matrix engine.
// bw[c] = 1 - (sp2 + popc(c) - 2*cross[c])/8  ->  argmin key(c) = popc(c) - 2*cross[c]
// A tile t (16x32 f16): rows = codes 16t..16t+15, K=0..7 = code bits; padded K
// slots are zero so f16 zero-padding is exact. D: lane n holds rows r, lane
// n+16 rows 8+r. Both rounds (B1 = positions of lanes 0..15, B2 = lanes 16..31)
// share each A tile and run interleaved so each round's select chain fills the
// other's WMMA->VALU hazard slots. popc(code) = popc(r+rowoff) [per-lane const]
// + popc(t) [wave-uniform SALU]. Codes ascend per chain, so strict '<' keeps
// the reference argmax-first tie-break; merges are lexicographic.
// ---------------------------------------------------------------------------
__device__ __forceinline__ void argmin256x2(v16h abase, v16h b1, v16h b2,
                                            float lowMask, int lane,
                                            int& code1, int& code2) {
    const int rowoff = (lane >= 16) ? 8 : 0;
    const _Float16 hz = (_Float16)0.0f;
    const _Float16 hm = (_Float16)lowMask;

    float bk[8];                         // popc of code bits 0..3 (per-lane const)
    #pragma unroll
    for (int r = 0; r < 8; ++r) bk[r] = (float)__popc(r + rowoff);

    float k0a = FLTBIG, k1a = FLTBIG, k0b = FLTBIG, k1b = FLTBIG;
    int   c0a = 0, c1a = 0, c0b = 0, c1b = 0;

    #pragma unroll 2
    for (int t = 0; t < 16; ++t) {
        v16h amat = abase;               // code bits 4..7 == bits of tile index t
        amat[4] = (t & 1) ? hm : hz;
        amat[5] = (t & 2) ? hm : hz;
        amat[6] = (t & 4) ? hm : hz;
        amat[7] = (t & 8) ? hm : hz;
        v8f da = {}, db = {};
        da = __builtin_amdgcn_wmma_f32_16x16x32_f16(
                 false, amat, false, b1, (short)0, da, false, false);
        db = __builtin_amdgcn_wmma_f32_16x16x32_f16(
                 false, amat, false, b2, (short)0, db, false, false);
        const float pt = (float)__popc(t);        // wave-uniform
        const int   cb = t * 16 + rowoff;
        #pragma unroll
        for (int r = 0; r < 4; ++r) {
            // chain 0: rows r, chain 1: rows r+4 (each ascending in code)
            const float base0 = bk[r] + pt;       // shared by both rounds
            const float base1 = bk[r + 4] + pt;
            const int   cA = cb + r;
            const int   cB = cb + r + 4;

            const float kAa = fmaf(-2.0f, da[r], base0);
            const bool  bAa = kAa < k0a;
            k0a = bAa ? kAa : k0a;  c0a = bAa ? cA : c0a;

            const float kAb = fmaf(-2.0f, db[r], base0);
            const bool  bAb = kAb < k0b;
            k0b = bAb ? kAb : k0b;  c0b = bAb ? cA : c0b;

            const float kBa = fmaf(-2.0f, da[r + 4], base1);
            const bool  bBa = kBa < k1a;
            k1a = bBa ? kBa : k1a;  c1a = bBa ? cB : c1a;

            const float kBb = fmaf(-2.0f, db[r + 4], base1);
            const bool  bBb = kBb < k1b;
            k1b = bBb ? kBb : k1b;  c1b = bBb ? cB : c1b;
        }
    }
    // merge the two chains per round (lexicographic -> argmax-first semantics)
    const bool ta = (k1a < k0a) | ((k1a == k0a) & (c1a < c0a));
    float mka = ta ? k1a : k0a;  int mca = ta ? c1a : c0a;
    const bool tb = (k1b < k0b) | ((k1b == k0b) & (c1b < c0b));
    float mkb = tb ? k1b : k0b;  int mcb = tb ? c1b : c0b;
    // merge the two half-wave candidate sets per round
    const float oka = __shfl_xor(mka, 16);
    const int   oca = __shfl_xor(mca, 16);
    const bool  sa  = (oka < mka) | ((oka == mka) & (oca < mca));
    code1 = sa ? oca : mca;
    const float okb = __shfl_xor(mkb, 16);
    const int   ocb = __shfl_xor(mcb, 16);
    const bool  sb  = (okb < mkb) | ((okb == mkb) & (ocb < mcb));
    code2 = sb ? ocb : mcb;
}

// ---------------------------------------------------------------------------
// accumulators in d_ws: acc[0]=num, acc[1]=den, acc[2]=vol_bce_sum
// ---------------------------------------------------------------------------
__global__ void SurfaceDice_init_kernel(float* __restrict__ acc) {
    if (threadIdx.x < 3 && blockIdx.x == 0) acc[threadIdx.x] = 0.0f;
}

__global__ __launch_bounds__(256)
__attribute__((amdgpu_waves_per_eu(4)))
void SurfaceDice_main_kernel(const float* __restrict__ pred,
                             const int*   __restrict__ labels,
                             const float* __restrict__ area,
                             float*       __restrict__ acc) {
    __shared__ float sArea[256];
    __shared__ float sRed[3][8];

    if (threadIdx.x < 256) sArea[threadIdx.x] = area[threadIdx.x];
    __syncthreads();

    const int lane = threadIdx.x & 31;
    const int wib  = threadIdx.x >> 5;
    const int wavesPerBlock = blockDim.x >> 5;
    const int gwave = blockIdx.x * wavesPerBlock + wib;
    const int nwave = gridDim.x * wavesPerBlock;

    const bool  lo      = (lane < 16);
    const float lowMask = lo ? 1.0f : 0.0f;

    // A-matrix static part: code bits 0..3 == bits of (lane&15); other slots zero.
    v16h abase = {};
    abase[0] = (_Float16)(((lane >> 0) & 1) ? lowMask : 0.0f);
    abase[1] = (_Float16)(((lane >> 1) & 1) ? lowMask : 0.0f);
    abase[2] = (_Float16)(((lane >> 2) & 1) ? lowMask : 0.0f);
    abase[3] = (_Float16)(((lane >> 3) & 1) ? lowMask : 0.0f);

    float accN = 0.0f, accD = 0.0f, accV = 0.0f;

    for (int g = gwave; g < NGROUPS32; g += nwave) {
        const int  p     = g * 32 + lane;
        const bool valid = (p < NPOS);

        // prefetch next grid-stride tile (speculative; dropped if OOB)
        {
            const int gp = g + nwave;
            if (gp < NGROUPS32) {
                const int pp = gp * 32 + lane;
                const int ip = pp / (385 * 385);
                const int rp = pp - ip * (385 * 385);
                const int yp = rp / 385;
                const int xp = rp - yp * 385;
                const int zc = min(max(ip - 1, 0), ZD - 1);
                const int yc = min(max(yp - 1, 0), YD - 1);
                const int xc = min(max(xp - 1, 0), XD - 1);
                const int idx = (zc * YD + yc) * XD + xc;
                __builtin_prefetch(pred + idx, 0, 3);
                __builtin_prefetch(labels + idx, 0, 3);
            }
        }

        // ---- per-lane scalar phase: every lane owns one position -----------
        V16HU mine;
        #pragma unroll
        for (int j = 0; j < 8; ++j) mine.u[j] = 0u;
        int   byte = 0;
        float sp2 = 0.0f, crossb = 0.0f, bces = 0.0f;

        if (valid) {
            const int i   = p / (385 * 385);
            const int rem = p - i * (385 * 385);
            const int y   = rem / 385;
            const int x   = rem - y * 385;
            float pcf[8];
            #pragma unroll
            for (int k = 0; k < 8; ++k) {
                const int dz = k >> 2, dy = (k >> 1) & 1, dx = k & 1;
                const int zr = i + dz - 1, yr = y + dy - 1, xr = x + dx - 1;
                const bool inb = ((unsigned)zr < (unsigned)ZD) &
                                 ((unsigned)yr < (unsigned)YD) &
                                 ((unsigned)xr < (unsigned)XD);
                // clamped index is always in-buffer -> unconditional load + select
                const int zc = min(max(zr, 0), ZD - 1);
                const int yc = min(max(yr, 0), YD - 1);
                const int xc = min(max(xr, 0), XD - 1);
                const int idx = (zc * YD + yc) * XD + xc;
                const float pv = pred[idx];
                const int   lv = labels[idx];
                const float pc = inb ? (1.0f / (1.0f + __expf(-pv))) : 0.0f;
                const int   lc = inb ? lv : 0;
                pcf[k]    = pc;
                byte     |= lc << k;
                sp2      += pc * pc;
                crossb   += lc ? pc : 0.0f;
                mine.h[k] = (_Float16)pc;
            }
            // masked BCE: interior (byte==0 || byte==255) is rare -> keep branch
            if (byte == 0 || byte == 255) {
                const float lf = (byte == 255) ? 1.0f : 0.0f;
                #pragma unroll
                for (int k = 0; k < 8; ++k) {
                    const float pc  = pcf[k];
                    const float lp  = __logf(fmaxf(pc, EPSLOG));
                    const float l1p = __logf(fmaxf(1.0f - pc, EPSLOG));
                    bces += -(lf * lp + (1.0f - lf) * l1p);
                }
            }
        }

        // partner halves for round 2 (all lanes execute the shuffles)
        const unsigned int pu0 = (unsigned)__shfl_xor((int)mine.u[0], 16);
        const unsigned int pu1 = (unsigned)__shfl_xor((int)mine.u[1], 16);
        const unsigned int pu2 = (unsigned)__shfl_xor((int)mine.u[2], 16);
        const unsigned int pu3 = (unsigned)__shfl_xor((int)mine.u[3], 16);

        // B1 columns = positions of lanes 0..15; B2 = lanes 16..31 (moved down)
        V16HU b1, b2;
        b1.u[0] = lo ? mine.u[0] : 0u;
        b1.u[1] = lo ? mine.u[1] : 0u;
        b1.u[2] = lo ? mine.u[2] : 0u;
        b1.u[3] = lo ? mine.u[3] : 0u;
        b1.u[4] = 0u; b1.u[5] = 0u; b1.u[6] = 0u; b1.u[7] = 0u;
        b2.u[0] = lo ? pu0 : 0u;
        b2.u[1] = lo ? pu1 : 0u;
        b2.u[2] = lo ? pu2 : 0u;
        b2.u[3] = lo ? pu3 : 0u;
        b2.u[4] = 0u; b2.u[5] = 0u; b2.u[6] = 0u; b2.u[7] = 0u;

        int code1, code2;
        argmin256x2(abase, b1.h, b2.h, lowMask, lane, code1, code2);

        const int mycode = lo ? code1 : code2;   // this lane's position's argmax code

        if (valid) {
            const float la  = sArea[byte];       // label_area
            const float pa  = sArea[mycode];     // pred_area
            const float nbb = (float)__popc(byte);
            const float pw  = 1.0f - (sp2 + nbb - 2.0f * crossb) * 0.125f; // exact f32
            accN += 2.0f * pw * la;
            accD += la + pa;
            accV += bces;
        }
    }

    // ---- wave -> block -> global reduction --------------------------------
    #pragma unroll
    for (int off = 16; off > 0; off >>= 1) {
        accN += __shfl_down(accN, off);
        accD += __shfl_down(accD, off);
        accV += __shfl_down(accV, off);
    }
    if (lane == 0) { sRed[0][wib] = accN; sRed[1][wib] = accD; sRed[2][wib] = accV; }
    __syncthreads();
    if (threadIdx.x == 0) {
        float n = 0.0f, d = 0.0f, v = 0.0f;
        for (int w = 0; w < wavesPerBlock; ++w) {
            n += sRed[0][w]; d += sRed[1][w]; v += sRed[2][w];
        }
        atomicAdd(&acc[0], n);
        atomicAdd(&acc[1], d);
        atomicAdd(&acc[2], v);
    }
}

__global__ void SurfaceDice_final_kernel(const float* __restrict__ acc,
                                         float* __restrict__ out) {
    if (threadIdx.x == 0 && blockIdx.x == 0) {
        const float num = acc[0], den = acc[1], vol = acc[2];
        const float dice = 1.0f - (num + SMOOTH) / (den + SMOOTH);
        out[0] = dice + vol * (1.0f / (8.0f * 148225.0f));  // mean over (8, 385*385)
    }
}

extern "C" void kernel_launch(void* const* d_in, const int* in_sizes, int n_in,
                              void* d_out, int out_size, void* d_ws, size_t ws_size,
                              hipStream_t stream) {
    const float* pred   = (const float*)d_in[0];
    const int*   labels = (const int*)d_in[1];
    const float* area   = (const float*)d_in[2];
    float*       acc    = (float*)d_ws;
    float*       out    = (float*)d_out;
    (void)in_sizes; (void)n_in; (void)out_size; (void)ws_size;

    SurfaceDice_init_kernel<<<1, 32, 0, stream>>>(acc);
    SurfaceDice_main_kernel<<<512, 256, 0, stream>>>(pred, labels, area, acc);
    SurfaceDice_final_kernel<<<1, 32, 0, stream>>>(acc, out);
}
